// MultiheadAttention_257698037892
// MI455X (gfx1250) — compile-verified, matrix operand order checked by reading, not run on hardware
//
#include <hip/hip_runtime.h>
#include <hip/hip_bf16.h>

// ---------------------------------------------------------------------------
// MI455X (gfx1250) fused multi-head attention, bf16 WMMA path.
//   - one-shot f32->bf16 prepack (halves resident set -> fits 192MB L2)
//   - GEMM: wave tile 32x64, 8 WMMAs per 32-wide K step, pure b128 loads
//   - flash attention: double-buffered LDS K / V^T tiles (async-copy K tile
//     via GLOBAL_LOAD_ASYNC_TO_LDS_B128 when the builtin exists), online
//     softmax with DPP butterfly row-max + ones-matrix WMMA row-sum
// ---------------------------------------------------------------------------

typedef __bf16 bf16_t;
typedef __attribute__((ext_vector_type(16))) __bf16 v16bf;
typedef __attribute__((ext_vector_type(8)))  __bf16 v8bf;
typedef __attribute__((ext_vector_type(8)))  float  v8f;
typedef __attribute__((ext_vector_type(4)))  int    v4i;

#define SEQ    4096
#define DMODEL 2048
#define NHEADS 16
#define DHEAD  128   // DMODEL / NHEADS

// ---------------------------------------------------------------------------
// Async global->LDS copy (CDNA5 GLOBAL_LOAD_ASYNC_TO_LDS_B128, ASYNCcnt).
// Builtin signature (from hipcc diagnostic): (v4i AS1*, v4i AS3*, imm, imm).
// Guarded: falls back to register staging when the builtin is unavailable.
// ---------------------------------------------------------------------------
#if __has_builtin(__builtin_amdgcn_global_load_async_to_lds_b128)
#define HAVE_ASYNC_COPY 1
typedef __attribute__((address_space(1))) v4i* gv4i_p;   // global v4i*
typedef __attribute__((address_space(3))) v4i* lv4i_p;   // LDS v4i*
static __device__ inline void async_copy16(const bf16_t* g, bf16_t* l) {
  __builtin_amdgcn_global_load_async_to_lds_b128((gv4i_p)g, (lv4i_p)l, 0, 0);
}
static __device__ inline void wait_async0() {
#if __has_builtin(__builtin_amdgcn_s_wait_asynccnt)
  __builtin_amdgcn_s_wait_asynccnt(0);
#else
  asm volatile("s_wait_asynccnt 0" ::: "memory");
#endif
}
#else
#define HAVE_ASYNC_COPY 0
#endif

// ---------------------------------------------------------------------------
// Fragment helpers (ISA 7.12.2 16-bit 16x32 A layout): lane L holds row L%16;
// half = L/16: elems 0..7 <-> K = k0+half*8+(0..7), elems 8..15 <-> +16.
// B operand uses the mirrored convention (lane holds col N=L%16).
// ---------------------------------------------------------------------------
static __device__ inline v16bf combine16(v8bf lo, v8bf hi) {
  v16bf a;
#pragma unroll
  for (int i = 0; i < 8; ++i) { a[i] = lo[i]; a[i + 8] = hi[i]; }
  return a;
}

static __device__ inline v16bf frag_from_bf16(const bf16_t* row, int k0, int half) {
  v8bf lo = *(const v8bf*)(row + k0 + half * 8);        // ds/global b128
  v8bf hi = *(const v8bf*)(row + k0 + 16 + half * 8);   // ds/global b128
  return combine16(lo, hi);
}

static __device__ inline v8f wmma_bf16(v16bf a, v16bf b, v8f c) {
  return __builtin_amdgcn_wmma_f32_16x16x32_bf16(false, a, false, b,
                                                 (short)0, c, false, false);
}

// ---------------------------------------------------------------------------
// DPP butterfly max all-reduce within each 16-lane group (lanes 0-15 / 16-31).
// xor1/xor2/xor4 via DPP8; "xor15" via DPP16 row_mirror. {1,2,4,15} spans
// GF(2)^4 -> full all-reduce. Pure VALU, DS pipe stays free for WMMA frags.
// ---------------------------------------------------------------------------
#define DPP8_SEL(a,b,c,d,e,f,g,h) \
  ((a)|((b)<<3)|((c)<<6)|((d)<<9)|((e)<<12)|((f)<<15)|((g)<<18)|((h)<<21))

static __device__ inline float bcast_max16(float x) {
  int i;
  i = __builtin_amdgcn_mov_dpp8(__float_as_int(x), DPP8_SEL(1,0,3,2,5,4,7,6));
  x = fmaxf(x, __int_as_float(i));
  i = __builtin_amdgcn_mov_dpp8(__float_as_int(x), DPP8_SEL(2,3,0,1,6,7,4,5));
  x = fmaxf(x, __int_as_float(i));
  i = __builtin_amdgcn_mov_dpp8(__float_as_int(x), DPP8_SEL(4,5,6,7,0,1,2,3));
  x = fmaxf(x, __int_as_float(i));
  i = __builtin_amdgcn_mov_dpp(__float_as_int(x), 0x140, 0xf, 0xf, true); // row_mirror
  x = fmaxf(x, __int_as_float(i));
  return x;
}

// ---------------------------------------------------------------------------
// Prepack: f32 -> bf16 (RNE), 8 elements per thread. One-shot, HBM-bound.
// ---------------------------------------------------------------------------
__global__ __launch_bounds__(256) void cvt_f32_bf16_x8(const float* __restrict__ src,
                                                       bf16_t* __restrict__ dst,
                                                       int n8) {
  const int i = blockIdx.x * 256 + threadIdx.x;
  if (i >= n8) return;
  const float4* s = (const float4*)src;
  float4 a = s[2 * i], b = s[2 * i + 1];
  v8bf o;
  o[0] = (bf16_t)a.x; o[1] = (bf16_t)a.y; o[2] = (bf16_t)a.z; o[3] = (bf16_t)a.w;
  o[4] = (bf16_t)b.x; o[5] = (bf16_t)b.y; o[6] = (bf16_t)b.z; o[7] = (bf16_t)b.w;
  *(v8bf*)(dst + (size_t)i * 8) = o;
}

// ---------------------------------------------------------------------------
// C = cscale * (A(MxK,bf16) @ W(NxK,bf16)^T). Block = 8 waves (4 Mrows x
// 2 Ncols); wave tile 32x64 (2x4 f32 accums); block tile 128x128; K step 32.
// ---------------------------------------------------------------------------
template <bool OUT_BF16>
__global__ __launch_bounds__(256) void gemm_nt_wmma(const bf16_t* __restrict__ A,
                                                    const bf16_t* __restrict__ W,
                                                    void* __restrict__ Cq,
                                                    int M, int N, int K,
                                                    float cscale) {
  const int lane = threadIdx.x & 31;
  const int wave = threadIdx.x >> 5;
  const int lm   = lane & 15;
  const int half = lane >> 4;
  const int wrow = wave >> 1;          // 0..3
  const int wcol = wave & 1;           // 0..1
  const int m0 = blockIdx.y * 128 + wrow * 32;
  const int n0 = blockIdx.x * 128 + wcol * 64;

  v8f acc[2][4] = {{{}, {}, {}, {}}, {{}, {}, {}, {}}};

  for (int k0 = 0; k0 < K; k0 += 32) {
    v16bf a[2], b[4];
#pragma unroll
    for (int i = 0; i < 2; ++i)
      a[i] = frag_from_bf16(A + (size_t)(m0 + i * 16 + lm) * K, k0, half);
#pragma unroll
    for (int j = 0; j < 4; ++j)
      b[j] = frag_from_bf16(W + (size_t)(n0 + j * 16 + lm) * K, k0, half);
#pragma unroll
    for (int i = 0; i < 2; ++i)
#pragma unroll
      for (int j = 0; j < 4; ++j)
        acc[i][j] = wmma_bf16(a[i], b[j], acc[i][j]);
  }

#pragma unroll
  for (int i = 0; i < 2; ++i)
#pragma unroll
    for (int j = 0; j < 4; ++j)
#pragma unroll
      for (int r = 0; r < 8; ++r) {
        const int row = m0 + i * 16 + half * 8 + r;
        const int col = n0 + j * 16 + lm;
        const float v = acc[i][j][r] * cscale;
        if (OUT_BF16)
          ((bf16_t*)Cq)[(size_t)row * N + col] = (bf16_t)v;
        else
          ((float*)Cq)[(size_t)row * N + col] = v;
      }
}

// ---------------------------------------------------------------------------
// Flash attention: grid = (SEQ/128, NHEADS); block = 256 thr = 8 waves; wave
// owns 16 query rows (Q pre-scaled by 1/sqrt(DHEAD) in the projection).
// Double-buffered 32-key K / V^T tiles; per step: QK^T (8 WMMAs), online
// softmax (DPP max, ones-WMMA row-sum), P via per-wave LDS, PV (8 WMMAs).
// ---------------------------------------------------------------------------
static __device__ inline void stageV_store(bf16_t* Vt, int skk, int sd0,
                                           v8bf v0, v8bf v1) {
#pragma unroll
  for (int i = 0; i < 8; ++i) {
    Vt[(sd0 + i) * 32 + skk]     = v0[i];
    Vt[(sd0 + 8 + i) * 32 + skk] = v1[i];
  }
}

__global__ __launch_bounds__(256) void flash_attn_wmma(const bf16_t* __restrict__ Qb,
                                                       const bf16_t* __restrict__ Kb,
                                                       const bf16_t* __restrict__ Vb,
                                                       bf16_t* __restrict__ On) {
  __shared__ bf16_t Ks[2][32 * DHEAD];    // K tiles [k][d]            (16 KB)
  __shared__ bf16_t Vts[2][DHEAD * 32];   // V^T tiles [d][k]          (16 KB)
  __shared__ bf16_t Ps[8 * 16 * 32];      // per-wave P staging         (8 KB)

  const int lane = threadIdx.x & 31;
  const int wave = threadIdx.x >> 5;
  const int lm   = lane & 15;
  const int half = lane >> 4;
  const int head = blockIdx.y;
  const int q0   = blockIdx.x * 128 + wave * 16;

  // staging coordinates: each thread moves one 16-element run of one key row
  const int skk = threadIdx.x >> 3;         // key row in tile, 0..31
  const int sd0 = (threadIdx.x & 7) * 16;   // d offset, 0..112

  const bf16_t* Qrow = Qb + (size_t)(q0 + lm) * DMODEL + head * DHEAD;
  v16bf qf[4];
#pragma unroll
  for (int c = 0; c < 4; ++c) qf[c] = frag_from_bf16(Qrow, c * 32, half);

  v16bf onesf;
#pragma unroll
  for (int i = 0; i < 16; ++i) onesf[i] = (bf16_t)1.0f;

  float mrow[8], lrow[8], alphav[8];
#pragma unroll
  for (int r = 0; r < 8; ++r) { mrow[r] = -1e30f; lrow[r] = 0.0f; }
  v8f o[8] = {{}, {}, {}, {}, {}, {}, {}, {}};

  bf16_t* Pw = Ps + wave * (16 * 32);
  const int NJB = SEQ / 32;

  // ---- prologue: stage tile 0 into buffer 0 ----
  {
    const bf16_t* vg = Vb + (size_t)skk * DMODEL + head * DHEAD + sd0;
    v8bf v0 = ((const v8bf*)vg)[0], v1 = ((const v8bf*)vg)[1];
    stageV_store(&Vts[0][0], skk, sd0, v0, v1);
    const bf16_t* kg = Kb + (size_t)skk * DMODEL + head * DHEAD + sd0;
#if HAVE_ASYNC_COPY
    async_copy16(kg,     &Ks[0][skk * DHEAD + sd0]);
    async_copy16(kg + 8, &Ks[0][skk * DHEAD + sd0 + 8]);
    wait_async0();
#else
    v8bf k0 = ((const v8bf*)kg)[0], k1 = ((const v8bf*)kg)[1];
    *(v8bf*)&Ks[0][skk * DHEAD + sd0]     = k0;
    *(v8bf*)&Ks[0][skk * DHEAD + sd0 + 8] = k1;
#endif
  }
  __syncthreads();

  for (int jb = 0; jb < NJB; ++jb) {
    const int buf = jb & 1, nb = buf ^ 1;
    const bool more = (jb + 1) < NJB;

    // ---- issue next tile's global traffic before computing ----
    v8bf nv0 = {}, nv1 = {}, nk0 = {}, nk1 = {};
    if (more) {
      const size_t gro = (size_t)((jb + 1) * 32 + skk) * DMODEL + head * DHEAD + sd0;
      const bf16_t* vg = Vb + gro;
      nv0 = ((const v8bf*)vg)[0]; nv1 = ((const v8bf*)vg)[1];
      const bf16_t* kg = Kb + gro;
#if HAVE_ASYNC_COPY
      async_copy16(kg,     &Ks[nb][skk * DHEAD + sd0]);       // overlaps compute
      async_copy16(kg + 8, &Ks[nb][skk * DHEAD + sd0 + 8]);
#else
      nk0 = ((const v8bf*)kg)[0]; nk1 = ((const v8bf*)kg)[1];
#endif
    }

    // ---- compute on current buffer ----
    const bf16_t* Kbuf = &Ks[buf][0];
    const bf16_t* Vbuf = &Vts[buf][0];

    // preload all 8 K fragments so ds_loads overlap, then the WMMA chain
    v16bf kf[2][4];
#pragma unroll
    for (int t = 0; t < 2; ++t)
#pragma unroll
      for (int c = 0; c < 4; ++c)
        kf[t][c] = frag_from_bf16(Kbuf + (t * 16 + lm) * DHEAD, c * 32, half);

    v8f s0 = {}, s1 = {};
#pragma unroll
    for (int c = 0; c < 4; ++c) {
      s0 = wmma_bf16(qf[c], kf[0][c], s0);
      s1 = wmma_bf16(qf[c], kf[1][c], s1);
    }

    // online softmax: DPP butterfly row-max, exp, rescale O, stash bf16 P
#pragma unroll
    for (int r = 0; r < 8; ++r) {
      const float x0 = s0[r], x1 = s1[r];
      const float mx = bcast_max16(fmaxf(x0, x1));
      const float mnew = fmaxf(mrow[r], mx);
      alphav[r] = __expf(mrow[r] - mnew);
      mrow[r] = mnew;
      const float p0 = __expf(x0 - mnew);
      const float p1 = __expf(x1 - mnew);
      Pw[(half * 8 + r) * 32 + lm]      = (bf16_t)p0;
      Pw[(half * 8 + r) * 32 + 16 + lm] = (bf16_t)p1;
#pragma unroll
      for (int t = 0; t < 8; ++t) o[t][r] *= alphav[r];
    }

    // within-wave DS ordering for the P staging round-trip
    asm volatile("s_wait_dscnt 0" ::: "memory");
    __builtin_amdgcn_wave_barrier();

    v16bf pf = frag_from_bf16(Pw + lm * 32, 0, half);   // A operand 16x32

    // row-sum of P via ones-matrix WMMA (sums exactly the bf16 P used in PV)
    v8f z = {};
    v8f lsum = wmma_bf16(pf, onesf, z);
#pragma unroll
    for (int r = 0; r < 8; ++r) lrow[r] = lrow[r] * alphav[r] + lsum[r];

    // preload all 8 V fragments, then the PV WMMA chain
    v16bf vfr[8];
#pragma unroll
    for (int t = 0; t < 8; ++t)
      vfr[t] = frag_from_bf16(Vbuf + (t * 16 + lm) * 32, 0, half);
#pragma unroll
    for (int t = 0; t < 8; ++t) o[t] = wmma_bf16(pf, vfr[t], o[t]);

    // ---- commit next tile's staging ----
    if (more) {
      stageV_store(&Vts[nb][0], skk, sd0, nv0, nv1);
#if !HAVE_ASYNC_COPY
      *(v8bf*)&Ks[nb][skk * DHEAD + sd0]     = nk0;
      *(v8bf*)&Ks[nb][skk * DHEAD + sd0 + 8] = nk1;
#endif
    }
#if HAVE_ASYNC_COPY
    if (more) wait_async0();
#endif
    __syncthreads();
  }

  // normalize and write concat-head output (col = head*128 + d), bf16
#pragma unroll
  for (int r = 0; r < 8; ++r) {
    const float inv = 1.0f / lrow[r];
#pragma unroll
    for (int t = 0; t < 8; ++t) {
      On[(size_t)(q0 + half * 8 + r) * DMODEL + head * DHEAD + t * 16 + lm] =
          (bf16_t)(o[t][r] * inv);
    }
  }
}

// ---------------------------------------------------------------------------
extern "C" void kernel_launch(void* const* d_in, const int* in_sizes, int n_in,
                              void* d_out, int out_size, void* d_ws, size_t ws_size,
                              hipStream_t stream) {
  (void)in_sizes; (void)n_in; (void)out_size; (void)ws_size;
  const float* query = (const float*)d_in[0];
  const float* key_  = (const float*)d_in[1];
  const float* value = (const float*)d_in[2];
  const float* Wq    = (const float*)d_in[3];
  const float* Wk    = (const float*)d_in[4];
  const float* Wv    = (const float*)d_in[5];
  const float* Wo    = (const float*)d_in[6];
  float* out = (float*)d_out;

  const size_t SD = (size_t)SEQ * DMODEL;     // 8.39M elems
  const size_t DD = (size_t)DMODEL * DMODEL;  // 4.19M elems

  // workspace (bf16): xq xk xv | qb kb vb | wq wk wv wo   (~134 MB)
  bf16_t* xq  = (bf16_t*)d_ws;        // reused as attn after Q projection
  bf16_t* xk  = xq + SD;
  bf16_t* xv  = xk + SD;
  bf16_t* qb  = xv + SD;
  bf16_t* kb  = qb + SD;
  bf16_t* vb  = kb + SD;
  bf16_t* wqb = vb + SD;
  bf16_t* wkb = wqb + DD;
  bf16_t* wvb = wkb + DD;
  bf16_t* wob = wvb + DD;
  bf16_t* attn = xq;  // xq is dead after the Q projection

  dim3 blk(256);

  const int sd8 = (int)(SD / 8), dd8 = (int)(DD / 8);
  cvt_f32_bf16_x8<<<dim3((sd8 + 255) / 256), blk, 0, stream>>>(query, xq, sd8);
  cvt_f32_bf16_x8<<<dim3((sd8 + 255) / 256), blk, 0, stream>>>(key_,  xk, sd8);
  cvt_f32_bf16_x8<<<dim3((sd8 + 255) / 256), blk, 0, stream>>>(value, xv, sd8);
  cvt_f32_bf16_x8<<<dim3((dd8 + 255) / 256), blk, 0, stream>>>(Wq, wqb, dd8);
  cvt_f32_bf16_x8<<<dim3((dd8 + 255) / 256), blk, 0, stream>>>(Wk, wkb, dd8);
  cvt_f32_bf16_x8<<<dim3((dd8 + 255) / 256), blk, 0, stream>>>(Wv, wvb, dd8);
  cvt_f32_bf16_x8<<<dim3((dd8 + 255) / 256), blk, 0, stream>>>(Wo, wob, dd8);

  dim3 ggemm(DMODEL / 128, SEQ / 128);  // (N tiles, M tiles)
  const float qscale = 0.08838834764831845f;  // 1/sqrt(DHEAD), folded into Q
  gemm_nt_wmma<true><<<ggemm, blk, 0, stream>>>(xq, wqb, qb, SEQ, DMODEL, DMODEL, qscale);
  gemm_nt_wmma<true><<<ggemm, blk, 0, stream>>>(xk, wkb, kb, SEQ, DMODEL, DMODEL, 1.0f);
  gemm_nt_wmma<true><<<ggemm, blk, 0, stream>>>(xv, wvb, vb, SEQ, DMODEL, DMODEL, 1.0f);

  dim3 gattn(SEQ / 128, NHEADS);
  flash_attn_wmma<<<gattn, blk, 0, stream>>>(qb, kb, vb, attn);

  gemm_nt_wmma<false><<<ggemm, blk, 0, stream>>>(attn, wob, out, SEQ, DMODEL, DMODEL, 1.0f);
}
